// Attention_51092930953251
// MI455X (gfx1250) — matrix operand order repaired and checked by reading, not compiled, not run on hardware
//
#include <hip/hip_runtime.h>
#include <stdint.h>

// ---------------------------------------------------------------------------
// Flash-attention (GQA) for MI455X / gfx1250, wave32 + WMMA f16 16x16x32.
// One workgroup per (b, h); 8 waves, each owning a 16-query-row slab.
// 64 keys per outer step; row-sums via ones-WMMA; row-max via ds_swizzle
// butterflies; fragment LDS loads software-pipelined around the WMMAs.
// ---------------------------------------------------------------------------

typedef __attribute__((ext_vector_type(16))) _Float16 v16h;
typedef __attribute__((ext_vector_type(8)))  float    v8f;
typedef __attribute__((ext_vector_type(8)))  uint32_t v8u;

#define B_     8
#define S_     128
#define DIM_   4096
#define KVDIM_ 1024
#define H_     32
#define KVH_   8
#define D_     128
#define MAXS_  2048
#define SCALE_ 0.08838834764831845f   // 1/sqrt(128)

#define TK     64    // keys per outer step
#define KSTR   136   // f16 row stride of K tile   (64 x 128, padded)
#define VSTR   68    // f16 row stride of V tile^T (128 x 64, padded)
#define PSTR   40    // f16 row stride of per-wave P buffer (16 x 32, padded)

__device__ __forceinline__ uint32_t pack2h(float x, float y) {
#if __has_builtin(__builtin_amdgcn_cvt_pkrtz)
    auto r = __builtin_amdgcn_cvt_pkrtz(x, y);   // v_cvt_pk_rtz_f16_f32
    return __builtin_bit_cast(uint32_t, r);
#else
    union { _Float16 h[2]; uint32_t u; } t;
    t.h[0] = (_Float16)x; t.h[1] = (_Float16)y;
    return t.u;
#endif
}

// butterfly xor within 16-lane halves using ds_swizzle (immediate pattern)
__device__ __forceinline__ float xor_max16(float x) {
    int r;
    r = __builtin_amdgcn_ds_swizzle(__builtin_bit_cast(int, x), (1 << 10) | 0x1f);
    x = fmaxf(x, __builtin_bit_cast(float, r));
    r = __builtin_amdgcn_ds_swizzle(__builtin_bit_cast(int, x), (2 << 10) | 0x1f);
    x = fmaxf(x, __builtin_bit_cast(float, r));
    r = __builtin_amdgcn_ds_swizzle(__builtin_bit_cast(int, x), (4 << 10) | 0x1f);
    x = fmaxf(x, __builtin_bit_cast(float, r));
    r = __builtin_amdgcn_ds_swizzle(__builtin_bit_cast(int, x), (8 << 10) | 0x1f);
    x = fmaxf(x, __builtin_bit_cast(float, r));
    return x;
}

__global__ __launch_bounds__(256)
void fa_gqa_kernel(const float* __restrict__ q,
                   const float* __restrict__ k,
                   const float* __restrict__ v,
                   const int*   __restrict__ startPtr,
                   const float* __restrict__ mask,
                   const float* __restrict__ cache_k,
                   const float* __restrict__ cache_v,
                   float*       __restrict__ out)
{
    __shared__ _Float16 kTile[TK][KSTR];     // [key][d]   17408 B
    __shared__ _Float16 vTileT[D_][VSTR];    // [d][key]   17408 B
    __shared__ _Float16 pBuf[8][16][PSTR];   // per-wave P 10240 B

    const int h    = blockIdx.x;          // 0..31 query head
    const int b    = blockIdx.y;          // 0..7  batch
    const int g    = h >> 2;              // KV head (n_rep = 4)
    const int tid  = threadIdx.x;
    const int wave = tid >> 5;
    const int lane = tid & 31;
    const int half = lane >> 4;
    const int lr   = lane & 15;

    const int start = startPtr[0];
    const int T     = start + S_;         // total keys; also mask row stride

    // ---- Q fragments: A-layout 16x32 f16, 4 chunks covering D=128 --------
    const int    qRow = wave * 16 + lr;
    const float* qrow = q + ((size_t)(b * S_ + qRow)) * DIM_ + h * D_;
    v16h qfrag[4];
    #pragma unroll
    for (int c = 0; c < 4; ++c) {
        v8u aq;
        #pragma unroll
        for (int j = 0; j < 4; ++j) {
            const int d0 = c * 32 + half * 16 + 2 * j;
            const int d1 = d0 + 8;
            aq[j]     = pack2h(qrow[d0], qrow[d0 + 1]);
            aq[4 + j] = pack2h(qrow[d1], qrow[d1 + 1]);
        }
        qfrag[c] = __builtin_bit_cast(v16h, aq);
    }

    v16h ones;                                 // B matrix of 1.0 for row-sums
    #pragma unroll
    for (int i = 0; i < 16; ++i) ones[i] = (_Float16)1.0f;

    // ---- softmax state: rows j + 8*half (C-layout) ------------------------
    float m[8];
    v8f   acc[8], accL;
    const v8f vzero = {};
    #pragma unroll
    for (int j = 0; j < 8; ++j) m[j] = -1e30f;
    #pragma unroll
    for (int n = 0; n < 8; ++n) acc[n] = vzero;
    accL = vzero;

    // ---- staging: thread covers 2 keys (row0, row0+1) x 16 d at sD0 -------
    const int row0 = (tid >> 3) * 2;
    const int sD0  = (tid & 7) * 16;
    uint32_t kreg[16], vreg[16];

    // running offsets (elements) for cache (coff) and input k/v (ioff)
    long coff = (((long)b * MAXS_ + row0) * KVH_ + g) * D_ + sD0;
    long ioff = ((long)b * S_ + (row0 - start)) * (long)KVDIM_ + g * D_ + sD0;
    const long cstep = (long)TK * KVH_ * D_;
    const long istep = (long)TK * KVDIM_;

    auto loadTileRegs = [&](int t0) {
        const int  ta = t0 + row0;
        const bool va_ = (ta < T), vb_ = (ta + 1 < T);
        const float* kpa = (ta     >= start) ? (k + ioff)          : (cache_k + coff);
        const float* kpb = (ta + 1 >= start) ? (k + ioff + KVDIM_) : (cache_k + coff + KVH_ * D_);
        const float* vpa = (ta     >= start) ? (v + ioff)          : (cache_v + coff);
        const float* vpb = (ta + 1 >= start) ? (v + ioff + KVDIM_) : (cache_v + coff + KVH_ * D_);
        float4 kqa[4], kqb[4], vqa[4], vqb[4];
        #pragma unroll
        for (int i4 = 0; i4 < 4; ++i4) {
            kqa[i4] = va_ ? ((const float4*)kpa)[i4] : float4{0,0,0,0};
            vqa[i4] = va_ ? ((const float4*)vpa)[i4] : float4{0,0,0,0};
            kqb[i4] = vb_ ? ((const float4*)kpb)[i4] : float4{0,0,0,0};
            vqb[i4] = vb_ ? ((const float4*)vpb)[i4] : float4{0,0,0,0};
        }
        #pragma unroll
        for (int i4 = 0; i4 < 4; ++i4) {
            kreg[2*i4]       = pack2h(kqa[i4].x, kqa[i4].y);
            kreg[2*i4 + 1]   = pack2h(kqa[i4].z, kqa[i4].w);
            kreg[8 + 2*i4]   = pack2h(kqb[i4].x, kqb[i4].y);
            kreg[8 + 2*i4+1] = pack2h(kqb[i4].z, kqb[i4].w);
            vreg[4*i4 + 0]   = pack2h(vqa[i4].x, vqb[i4].x);  // key pair at d
            vreg[4*i4 + 1]   = pack2h(vqa[i4].y, vqb[i4].y);
            vreg[4*i4 + 2]   = pack2h(vqa[i4].z, vqb[i4].z);
            vreg[4*i4 + 3]   = pack2h(vqa[i4].w, vqb[i4].w);
        }
    };

    loadTileRegs(0);   // prologue: tile 0 into registers

    const int nsteps = (T + TK - 1) / TK;
    for (int stepi = 0; stepi < nsteps; ++stepi) {
        const int t0 = stepi * TK;

        __syncthreads();   // all waves finished reading previous tile

        // ---- store staged registers into LDS (all b32 stores) -------------
        {
            uint32_t* kr0 = (uint32_t*)&kTile[row0][0];
            uint32_t* kr1 = (uint32_t*)&kTile[row0 + 1][0];
            #pragma unroll
            for (int i = 0; i < 8; ++i) {
                kr0[sD0 / 2 + i] = kreg[i];
                kr1[sD0 / 2 + i] = kreg[8 + i];
            }
            #pragma unroll
            for (int i = 0; i < 16; ++i)
                *(uint32_t*)&vTileT[sD0 + i][row0] = vreg[i];
        }
        __syncthreads();

        // ---- issue global loads for the NEXT tile (overlaps compute) ------
        coff += cstep; ioff += istep;
        if (stepi + 1 < nsteps) loadTileRegs(t0 + TK);

        // ---- scores: 4 key-subtiles x 4 d-chunks, pipelined B frags -------
        v8f sc[4] = {vzero, vzero, vzero, vzero};
        {
            auto loadKfrag = [&](int idx) -> v8u {
                const int n = idx >> 2, c = idx & 3;
                const uint32_t* krow = (const uint32_t*)&kTile[n * 16 + lr][0];
                v8u bq;
                #pragma unroll
                for (int j = 0; j < 8; ++j)
                    bq[j] = krow[c * 16 + half * 8 + j];
                return bq;
            };
            v8u cur = loadKfrag(0);
            #pragma unroll
            for (int idx = 0; idx < 16; ++idx) {
                v8u nxt = cur;
                if (idx < 15) nxt = loadKfrag(idx + 1);
                const int n = idx >> 2, c = idx & 3;
                v16h bfrag = __builtin_bit_cast(v16h, cur);
                sc[n] = __builtin_amdgcn_wmma_f32_16x16x32_f16(
                    false, qfrag[c], false, bfrag, (short)0, sc[n], false, false);
                cur = nxt;
            }
        }

        // ---- mask + online softmax (only row-MAX crosses lanes) -----------
        #pragma unroll
        for (int j = 0; j < 8; ++j) {
            const int srow = wave * 16 + half * 8 + j;
            const float* mrow = mask + (size_t)srow * T;
            #pragma unroll
            for (int n = 0; n < 4; ++n) {
                const int t = t0 + n * 16 + lr;
                const float mv = (t < T) ? mrow[t] : -1e30f;
                sc[n][j] = sc[n][j] * SCALE_ + mv;
            }
            float mx = fmaxf(fmaxf(sc[0][j], sc[1][j]),
                             fmaxf(sc[2][j], sc[3][j]));
            mx = xor_max16(mx);
            const float nm   = fmaxf(m[j], mx);
            const float corr = __expf(m[j] - nm);
            m[j] = nm;
            #pragma unroll
            for (int n = 0; n < 4; ++n)
                sc[n][j] = __expf(sc[n][j] - nm);
            accL[j] *= corr;
            #pragma unroll
            for (int n = 0; n < 8; ++n) acc[n][j] *= corr;
        }

        // ---- PV in two 32-key halves, pipelined V frags -------------------
        #pragma unroll
        for (int hh = 0; hh < 2; ++hh) {
            #pragma unroll
            for (int j = 0; j < 8; ++j) {
                const int prow = half * 8 + j;
                pBuf[wave][prow][lr]      = (_Float16)sc[2 * hh][j];
                pBuf[wave][prow][16 + lr] = (_Float16)sc[2 * hh + 1][j];
            }
            v8u pq;
            const uint32_t* pr = (const uint32_t*)&pBuf[wave][lr][0];
            #pragma unroll
            for (int j = 0; j < 4; ++j) {
                pq[j]     = pr[half * 4 + j];
                pq[4 + j] = pr[8 + half * 4 + j];
            }
            v16h pfrag = __builtin_bit_cast(v16h, pq);

            auto loadVfrag = [&](int n) -> v8u {
                const uint32_t* vrow = (const uint32_t*)&vTileT[n * 16 + lr][0];
                v8u bq;
                #pragma unroll
                for (int j = 0; j < 8; ++j)
                    bq[j] = vrow[hh * 16 + (j >> 2) * 8 + half * 4 + (j & 3)];
                return bq;
            };
            v8u cur = loadVfrag(0);
            #pragma unroll
            for (int n = 0; n < 8; ++n) {
                v8u nxt = cur;
                if (n < 7) nxt = loadVfrag(n + 1);
                v16h vfrag = __builtin_bit_cast(v16h, cur);
                acc[n] = __builtin_amdgcn_wmma_f32_16x16x32_f16(
                    false, pfrag, false, vfrag, (short)0, acc[n], false, false);
                cur = nxt;
            }
            accL = __builtin_amdgcn_wmma_f32_16x16x32_f16(
                false, pfrag, false, ones, (short)0, accL, false, false);
        }
    }

    // ---- normalize by accL (row-sum) and write out[b, s, h*D + d] ---------
    #pragma unroll
    for (int j = 0; j < 8; ++j) {
        const float inv  = 1.0f / accL[j];
        const int   srow = wave * 16 + half * 8 + j;
        float* orow = out + ((size_t)(b * S_ + srow)) * DIM_ + h * D_;
        #pragma unroll
        for (int n = 0; n < 8; ++n)
            orow[n * 16 + lr] = acc[n][j] * inv;
    }
}

extern "C" void kernel_launch(void* const* d_in, const int* in_sizes, int n_in,
                              void* d_out, int out_size, void* d_ws, size_t ws_size,
                              hipStream_t stream) {
    (void)in_sizes; (void)n_in; (void)out_size; (void)d_ws; (void)ws_size;
    const float* q     = (const float*)d_in[0];
    const float* k     = (const float*)d_in[1];
    const float* v     = (const float*)d_in[2];
    const int*   start = (const int*)  d_in[3];
    // d_in[4] = freqs_cis (unused by the reference)
    const float* mask  = (const float*)d_in[5];
    const float* ck    = (const float*)d_in[6];
    const float* cv    = (const float*)d_in[7];

    dim3 grid(H_, B_);     // one workgroup per (head, batch)
    fa_gqa_kernel<<<grid, 256, 0, stream>>>(q, k, v, start, mask, ck, cv,
                                            (float*)d_out);
}
